// DeformableAlignment_14302241095861
// MI455X (gfx1250) — matrix operand order repaired and checked by reading, not compiled, not run on hardware
//
#include <hip/hip_runtime.h>
#include <hip/hip_bf16.h>
#include <stdint.h>

typedef __attribute__((ext_vector_type(16))) _Float16     v16h;
typedef __attribute__((ext_vector_type(8)))  float        v8f;
typedef __attribute__((ext_vector_type(4)))  unsigned int u32x4;

union HFrag { v16h h; u32x4 u[2]; };

// ---------------------------------------------------------------------------
// Constants for this problem instance
// ---------------------------------------------------------------------------
#define HH   192
#define WW   192
#define HWC  (HH * WW)          // 36864
#define NB   2                  // batch
#define CC   64                 // feature channels
#define NPIX (NB * HWC)         // 73728 total pixels (divisible by 128)
#define BN   128                // pixels per block tile

// low 32 bits of a generic pointer to __shared__ == wave-relative LDS address
__device__ __forceinline__ unsigned int lds_addr32(const void* p) {
    return (unsigned int)(uintptr_t)p;
}

// async contiguous 16B copy global -> LDS (GLOBAL_LOAD_ASYNC_TO_LDS_B128,
// tracked with ASYNCcnt). saddr = off (null), GV addressing.
__device__ __forceinline__ void async_copy_b128(unsigned int lds_off,
                                                const void* gptr) {
    asm volatile("global_load_async_to_lds_b128 %0, %1, off"
                 :
                 : "v"(lds_off), "v"((unsigned long long)(uintptr_t)gptr)
                 : "memory");
}
__device__ __forceinline__ void wait_async0() {
    asm volatile("s_wait_asynccnt 0x0" ::: "memory");
}

// ---------------------------------------------------------------------------
// Kernel 0: weight pack  fp32 [Cout][Ktot] -> f16 [Mpad=64][Kpad], zero padded
// ---------------------------------------------------------------------------
__global__ void pack_weights(const float* __restrict__ w, _Float16* __restrict__ o,
                             int Cout, int Ktot, int Kpad) {
    int i = blockIdx.x * 256 + threadIdx.x;
    int tot = 64 * Kpad;
    if (i >= tot) return;
    int m = i / Kpad;
    int k = i - m * Kpad;
    float v = (m < Cout && k < Ktot) ? w[m * Ktot + k] : 0.0f;
    o[i] = (_Float16)v;
}

// ---------------------------------------------------------------------------
// Kernel 1: implicit-GEMM 3x3 conv with WMMA f16->f32.
//   Block: 256 threads (8 waves). Tile: 64 out-ch x 128 pixels. K chunk = 32.
//   Each wave: 1 A fragment (16x32) reused across 4 B fragments -> 4 WMMAs.
//   ACT: 0 = none, 1 = leaky-relu(0.1)
// ---------------------------------------------------------------------------
template <typename InT, typename OutT, int ACT>
__global__ __launch_bounds__(256) void conv3x3_wmma(
    const InT* __restrict__ in, const _Float16* __restrict__ wq,
    const float* __restrict__ bias, OutT* __restrict__ out,
    int Cin, int Cout, int Ktot, int Kpad) {

    __shared__ _Float16 At[64 * 32];    // A tile: [m][k]
    __shared__ _Float16 Bt[BN * 32];    // B tile: [n][k] (K contiguous per pixel)

    const int tid  = threadIdx.x;
    const int lane = tid & 31;
    const int wv   = tid >> 5;
    const int wm   = wv & 3;            // wave M sub-tile (16 rows)
    const int wn   = wv >> 2;           // wave N sub-tile (64 pixels)
    const int blockN = blockIdx.x * BN;

    // A-tile staging coords (8 halves = 16B per thread)
    const int arow = tid >> 2;          // 0..63
    const int akh  = (tid & 3) * 8;     // 0,8,16,24
    const unsigned int aLds = lds_addr32(&At[arow * 32 + akh]);
    const _Float16* gA = wq + (size_t)arow * Kpad + akh;

    // B-tile staging coords (16 halves per thread)
    const int frow = tid >> 1;          // 0..127 : pixel row in tile
    const int fkh  = (tid & 1) * 16;    // 0,16
    const int p    = blockN + frow;
    const int pb   = p / HWC;
    const int pr   = p - pb * HWC;
    const int py   = pr / WW;
    const int px   = pr - py * WW;
    const InT* inB = in + (size_t)pb * Cin * HWC;

    v8f acc[4] = {};

    for (int k0 = 0; k0 < Kpad; k0 += 32) {
        // ---- A tile: async global->LDS copy, overlapped with im2col below ----
        async_copy_b128(aLds, gA + k0);

        // ---- B tile: on-the-fly im2col with zero padding ----
        #pragma unroll
        for (int j = 0; j < 16; ++j) {
            int k = k0 + fkh + j;
            float v = 0.0f;
            if (k < Ktot) {
                int c  = k / 9;
                int kk = k - c * 9;
                int ky = kk / 3;
                int kx = kk - ky * 3;
                int sy = py + ky - 1;
                int sx = px + kx - 1;
                if (sy >= 0 && sy < HH && sx >= 0 && sx < WW)
                    v = (float)inB[(size_t)c * HWC + sy * WW + sx];
            }
            Bt[frow * 32 + fkh + j] = (_Float16)v;
        }
        wait_async0();
        __syncthreads();

        // ---- per-lane fragment loads (wave32 WMMA layout, ISA 7.12.2) ----
        HFrag a;
        const int mrow = wm * 16 + (lane & 15);
        const int o1   = (lane >> 4) * 8;
        a.u[0] = *reinterpret_cast<const u32x4*>(&At[mrow * 32 + o1]);
        a.u[1] = *reinterpret_cast<const u32x4*>(&At[mrow * 32 + 16 + o1]);

        const int kb = (lane >> 4) * 16;
        #pragma unroll
        for (int s = 0; s < 4; ++s) {
            HFrag b;
            const int n = wn * 64 + s * 16 + (lane & 15);
            b.u[0] = *reinterpret_cast<const u32x4*>(&Bt[n * 32 + o1]);
            b.u[1] = *reinterpret_cast<const u32x4*>(&Bt[n * 32 + 16 + o1]);
            acc[s] = __builtin_amdgcn_wmma_f32_16x16x32_f16(
                false, a.h, false, b.h, (short)0, acc[s], false, false);
        }
        __syncthreads();
    }

    // ---- epilogue: bias + activation + NCHW store ----
    const int mbase = wm * 16 + 8 * (lane >> 4);
    #pragma unroll
    for (int r = 0; r < 8; ++r) {
        int m = mbase + r;
        if (m >= Cout) continue;
        float bv = bias[m];
        #pragma unroll
        for (int s = 0; s < 4; ++s) {
            int nloc = wn * 64 + s * 16 + (lane & 15);
            int pp   = blockN + nloc;
            int bb   = pp / HWC;
            int rr   = pp - bb * HWC;
            float vsum = acc[s][r] + bv;
            if (ACT) vsum = (vsum >= 0.0f) ? vsum : 0.1f * vsum;
            out[(size_t)bb * Cout * HWC + (size_t)m * HWC + rr] = (OutT)vsum;
        }
    }
}

// ---------------------------------------------------------------------------
// Kernel 2: offset/mask finalize.
//   offset[ch] = 10*tanh(raw[ch]) + flow_yx(ch%2),  ch in [0,18)
//   mask[k]    = sigmoid(raw[18+k])
// ---------------------------------------------------------------------------
__global__ void finalize_offsets(const float* __restrict__ raw,
                                 const float* __restrict__ flow,
                                 float* __restrict__ off,
                                 float* __restrict__ msk) {
    int i = blockIdx.x * 256 + threadIdx.x;     // over NB*9*HWC
    const int total = NB * 9 * HWC;
    if (i >= total) return;
    int b = i / (9 * HWC);
    int t = i - b * 9 * HWC;
    int k = t / HWC;
    int r = t - k * HWC;

    const float* rb = raw + (size_t)b * 27 * HWC;
    float fy = flow[((size_t)b * 2 + 1) * HWC + r];
    float fx = flow[((size_t)b * 2 + 0) * HWC + r];

    float dy = 10.0f * tanhf(rb[(size_t)(2 * k)     * HWC + r]) + fy;
    float dx = 10.0f * tanhf(rb[(size_t)(2 * k + 1) * HWC + r]) + fx;
    float mvraw = rb[(size_t)(18 + k) * HWC + r];
    float mv = 1.0f / (1.0f + expf(-mvraw));

    off[((size_t)b * 18 + 2 * k)     * HWC + r] = dy;
    off[((size_t)b * 18 + 2 * k + 1) * HWC + r] = dx;
    msk[((size_t)b * 9 + k) * HWC + r]          = mv;
}

// ---------------------------------------------------------------------------
// Kernel 3: deformable conv as implicit GEMM (same tiling as conv kernel).
//   Per block: precompute 128x9 bilinear taps (weights fold mask+validity)
//   in LDS once, then 18 K-chunks of the WMMA pipeline.
// ---------------------------------------------------------------------------
struct BilinPre { float w00, w01, w10, w11; int i00, i01, i10, i11; };

__global__ __launch_bounds__(256) void deform_wmma(
    const float* __restrict__ x, const float* __restrict__ off,
    const float* __restrict__ msk, const _Float16* __restrict__ wq,
    const float* __restrict__ bias, float* __restrict__ out) {

    constexpr int Kpad = CC * 9;        // 576 = 18 * 32, exact
    __shared__ BilinPre pre[BN * 9];    // 36 KB
    __shared__ _Float16 At[64 * 32];
    __shared__ _Float16 Bt[BN * 32];

    const int tid  = threadIdx.x;
    const int lane = tid & 31;
    const int wv   = tid >> 5;
    const int wm   = wv & 3;
    const int wn   = wv >> 2;
    const int blockN = blockIdx.x * BN;

    // ---- precompute bilinear taps for this block's 128 pixels x 9 taps ----
    for (int e = tid; e < BN * 9; e += 256) {
        int prow = e / 9;
        int kk   = e - prow * 9;
        int p = blockN + prow;
        int b = p / HWC;
        int r = p - b * HWC;
        int yy = r / WW;
        int xx = r - yy * WW;

        float dy = off[((size_t)b * 18 + 2 * kk)     * HWC + r];
        float dx = off[((size_t)b * 18 + 2 * kk + 1) * HWC + r];
        float mk = msk[((size_t)b * 9 + kk) * HWC + r];

        float pyf = (float)(yy - 1 + kk / 3) + dy;
        float pxf = (float)(xx - 1 + kk % 3) + dx;
        float y0f = floorf(pyf);
        float x0f = floorf(pxf);
        float ly = pyf - y0f;
        float lx = pxf - x0f;
        int y0 = (int)y0f, x0i = (int)x0f;
        int y1 = y0 + 1,   x1i = x0i + 1;

        float v00 = (y0 >= 0 && y0 < HH && x0i >= 0 && x0i < WW) ? 1.0f : 0.0f;
        float v01 = (y0 >= 0 && y0 < HH && x1i >= 0 && x1i < WW) ? 1.0f : 0.0f;
        float v10 = (y1 >= 0 && y1 < HH && x0i >= 0 && x0i < WW) ? 1.0f : 0.0f;
        float v11 = (y1 >= 0 && y1 < HH && x1i >= 0 && x1i < WW) ? 1.0f : 0.0f;

        int yc0 = y0 < 0 ? 0 : (y0 > HH - 1 ? HH - 1 : y0);
        int yc1 = y1 < 0 ? 0 : (y1 > HH - 1 ? HH - 1 : y1);
        int xc0 = x0i < 0 ? 0 : (x0i > WW - 1 ? WW - 1 : x0i);
        int xc1 = x1i < 0 ? 0 : (x1i > WW - 1 ? WW - 1 : x1i);

        BilinPre q;
        q.w00 = (1.0f - ly) * (1.0f - lx) * mk * v00;
        q.w01 = (1.0f - ly) * lx          * mk * v01;
        q.w10 = ly          * (1.0f - lx) * mk * v10;
        q.w11 = ly          * lx          * mk * v11;
        q.i00 = yc0 * WW + xc0;
        q.i01 = yc0 * WW + xc1;
        q.i10 = yc1 * WW + xc0;
        q.i11 = yc1 * WW + xc1;
        pre[e] = q;
    }

    // A-tile staging coords
    const int arow = tid >> 2;
    const int akh  = (tid & 3) * 8;
    const unsigned int aLds = lds_addr32(&At[arow * 32 + akh]);
    const _Float16* gA = wq + (size_t)arow * Kpad + akh;

    // B-tile staging coords
    const int frow = tid >> 1;          // pixel row (0..127)
    const int fkh  = (tid & 1) * 16;
    const int p    = blockN + frow;
    const int pb   = p / HWC;
    const float* xB = x + (size_t)pb * CC * HWC;

    v8f acc[4] = {};
    __syncthreads();   // pre[] ready

    for (int k0 = 0; k0 < Kpad; k0 += 32) {
        async_copy_b128(aLds, gA + k0);

        #pragma unroll
        for (int j = 0; j < 16; ++j) {
            int k  = k0 + fkh + j;
            int c  = k / 9;
            int kk = k - c * 9;
            BilinPre q = pre[frow * 9 + kk];
            const float* xc = xB + (size_t)c * HWC;
            float v = q.w00 * xc[q.i00] + q.w01 * xc[q.i01] +
                      q.w10 * xc[q.i10] + q.w11 * xc[q.i11];
            Bt[frow * 32 + fkh + j] = (_Float16)v;
        }
        wait_async0();
        __syncthreads();

        HFrag a;
        const int mrow = wm * 16 + (lane & 15);
        const int o1   = (lane >> 4) * 8;
        a.u[0] = *reinterpret_cast<const u32x4*>(&At[mrow * 32 + o1]);
        a.u[1] = *reinterpret_cast<const u32x4*>(&At[mrow * 32 + 16 + o1]);

        const int kb = (lane >> 4) * 16;
        #pragma unroll
        for (int s = 0; s < 4; ++s) {
            HFrag b;
            const int n = wn * 64 + s * 16 + (lane & 15);
            b.u[0] = *reinterpret_cast<const u32x4*>(&Bt[n * 32 + o1]);
            b.u[1] = *reinterpret_cast<const u32x4*>(&Bt[n * 32 + 16 + o1]);
            acc[s] = __builtin_amdgcn_wmma_f32_16x16x32_f16(
                false, a.h, false, b.h, (short)0, acc[s], false, false);
        }
        __syncthreads();
    }

    // ---- epilogue: bias + f32 NCHW store ----
    const int mbase = wm * 16 + 8 * (lane >> 4);
    #pragma unroll
    for (int r = 0; r < 8; ++r) {
        int m = mbase + r;
        float bv = bias[m];
        #pragma unroll
        for (int s = 0; s < 4; ++s) {
            int nloc = wn * 64 + s * 16 + (lane & 15);
            int pp   = blockN + nloc;
            int bb   = pp / HWC;
            int rr   = pp - bb * HWC;
            out[(size_t)bb * CC * HWC + (size_t)m * HWC + rr] = acc[s][r] + bv;
        }
    }
}

// ---------------------------------------------------------------------------
// Host launcher
// ---------------------------------------------------------------------------
extern "C" void kernel_launch(void* const* d_in, const int* in_sizes, int n_in,
                              void* d_out, int out_size, void* d_ws, size_t ws_size,
                              hipStream_t stream) {
    (void)in_sizes; (void)n_in; (void)out_size; (void)ws_size;

    const float* x    = (const float*)d_in[0];
    const float* cond = (const float*)d_in[1];
    const float* flow = (const float*)d_in[2];
    const float* w1 = (const float*)d_in[3];  const float* b1 = (const float*)d_in[4];
    const float* w2 = (const float*)d_in[5];  const float* b2 = (const float*)d_in[6];
    const float* w3 = (const float*)d_in[7];  const float* b3 = (const float*)d_in[8];
    const float* w4 = (const float*)d_in[9];  const float* b4 = (const float*)d_in[10];
    const float* wd = (const float*)d_in[11]; const float* bd = (const float*)d_in[12];
    float* out = (float*)d_out;

    constexpr int K1  = 131 * 9;     // 1179
    constexpr int K1p = 1184;        // 37 * 32
    constexpr int K2  = 576;         // 18 * 32 exactly

    // bump allocator over workspace (256B aligned regions)
    char* ws = (char*)d_ws;
    size_t cur = 0;
    auto alloc = [&](size_t bytes) -> void* {
        void* pp = ws + cur;
        cur += (bytes + 255) & ~(size_t)255;
        return pp;
    };
    _Float16* wp1 = (_Float16*)alloc((size_t)64 * K1p * 2);
    _Float16* wp2 = (_Float16*)alloc((size_t)64 * K2 * 2);
    _Float16* wp3 = (_Float16*)alloc((size_t)64 * K2 * 2);
    _Float16* wp4 = (_Float16*)alloc((size_t)64 * K2 * 2);
    _Float16* wp5 = (_Float16*)alloc((size_t)64 * K2 * 2);
    _Float16* h1  = (_Float16*)alloc((size_t)NB * CC * HWC * 2);
    _Float16* h2  = (_Float16*)alloc((size_t)NB * CC * HWC * 2);
    float* raw4   = (float*)alloc((size_t)NB * 27 * HWC * 4);
    float* offb   = (float*)alloc((size_t)NB * 18 * HWC * 4);
    float* mskb   = (float*)alloc((size_t)NB * 9 * HWC * 4);

    // weight packing
    pack_weights<<<(64 * K1p + 255) / 256, 256, 0, stream>>>(w1, wp1, 64, K1, K1p);
    pack_weights<<<(64 * K2  + 255) / 256, 256, 0, stream>>>(w2, wp2, 64, K2, K2);
    pack_weights<<<(64 * K2  + 255) / 256, 256, 0, stream>>>(w3, wp3, 64, K2, K2);
    pack_weights<<<(64 * K2  + 255) / 256, 256, 0, stream>>>(w4, wp4, 27, K2, K2);
    pack_weights<<<(64 * K2  + 255) / 256, 256, 0, stream>>>(wd, wp5, 64, K2, K2);

    dim3 grid(NPIX / BN);   // 576 pixel tiles

    // conv stack
    conv3x3_wmma<float,    _Float16, 1><<<grid, 256, 0, stream>>>(cond, wp1, b1, h1,   131, 64, K1, K1p);
    conv3x3_wmma<_Float16, _Float16, 1><<<grid, 256, 0, stream>>>(h1,   wp2, b2, h2,    64, 64, K2, K2);
    conv3x3_wmma<_Float16, _Float16, 1><<<grid, 256, 0, stream>>>(h2,   wp3, b3, h1,    64, 64, K2, K2);
    conv3x3_wmma<_Float16, float,    0><<<grid, 256, 0, stream>>>(h1,   wp4, b4, raw4,  64, 27, K2, K2);

    // offsets / mask
    finalize_offsets<<<(NB * 9 * HWC + 255) / 256, 256, 0, stream>>>(raw4, flow, offb, mskb);

    // deformable conv
    deform_wmma<<<grid, 256, 0, stream>>>(x, offb, mskb, wp5, bd, out);
}
